// AmharicHNetMixer_63917703299658
// MI455X (gfx1250) — compile-verified, matrix operand order checked by reading, not run on hardware
//
#include <hip/hip_runtime.h>
#include <hip/hip_bf16.h>

// ---------------------------------------------------------------------------
// Model dims
// ---------------------------------------------------------------------------
constexpr int kD   = 512;
constexpr int kDFF = 2048;
constexpr int kV   = 256;
constexpr int kNL  = 6;
constexpr int kH   = 8;
constexpr int kC   = 128;
constexpr int kB   = 16;
constexpr int kL   = 2048;
constexpr float kEPS = 1e-5f;

typedef __attribute__((ext_vector_type(16))) __bf16 v16bf;
typedef __attribute__((ext_vector_type(8)))  __bf16 v8bf;
typedef __attribute__((ext_vector_type(8)))  float  v8f;

__device__ __forceinline__ unsigned short f2bf(float f) {
  unsigned int u = __float_as_uint(f);
  u += 0x7FFFu + ((u >> 16) & 1u);   // round-to-nearest-even
  return (unsigned short)(u >> 16);
}

// ---------------------------------------------------------------------------
// Embedding + sinusoidal positional encoding -> fp32 + bf16
// ---------------------------------------------------------------------------
__global__ void embed_pos_kernel(const int* __restrict__ ids,
                                 const float* __restrict__ emb,
                                 float* __restrict__ x,
                                 unsigned short* __restrict__ xbf) {
  const int bl  = blockIdx.x;          // 0..B*L-1
  const int l   = bl % kL;
  const int tok = ids[bl];
  for (int d = threadIdx.x; d < kD; d += 256) {
    const int h = d >> 1;
    const float div = __expf((float)(2 * h) * (-9.210340371976184f / (float)kD));
    const float ang = (float)l * div;
    const float pe  = (d & 1) ? __cosf(ang) : __sinf(ang);
    const float v   = emb[tok * kD + d] + pe;
    const size_t idx = (size_t)bl * kD + d;
    x[idx]   = v;
    xbf[idx] = f2bf(v);
  }
}

// binp[b,l, 0:512] = x[b,l], binp[b,l,512:1024] = x[b,l+1] (0 for last)
__global__ void build_binp_kernel(const unsigned short* __restrict__ xbf,
                                  unsigned short* __restrict__ binp) {
  const size_t i = (size_t)blockIdx.x * 256 + threadIdx.x;  // over B*L*D
  if (i >= (size_t)kB * kL * kD) return;
  const int d = (int)(i % kD);
  const size_t bl = i / kD;
  const int l = (int)(bl % kL);
  binp[bl * (2 * kD) + d]      = xbf[i];
  binp[bl * (2 * kD) + kD + d] = (l + 1 < kL) ? xbf[i + kD] : (unsigned short)0;
}

// ---------------------------------------------------------------------------
// Weight transpose + fp32 -> bf16 : Wt[n*K+k] = bf16(W[k*N+n])
// ---------------------------------------------------------------------------
__global__ void conv_transpose_kernel(const float* __restrict__ W,
                                      unsigned short* __restrict__ Wt,
                                      int K, int N) {
  const size_t i = (size_t)blockIdx.x * 256 + threadIdx.x;
  if (i >= (size_t)K * N) return;
  const int n = (int)(i / K);
  const int k = (int)(i % K);
  Wt[i] = f2bf(W[(size_t)k * N + n]);
}

// ---------------------------------------------------------------------------
// WMMA bf16 GEMM with async global->LDS double-buffered staging.
//   C[M,N] = A[M,K] @ Bt[N,K]^T + bias, optional relu / bf16 copy.
// Block: 256 threads = 8 waves. Block tile: 128(M) x 64(N), K-chunk 64.
// Stage copy: A-tile 16KB + B-tile 8KB = 6 async B128 instructions per wave.
// Pipeline: issue stage k+1, s_wait_asynccnt 6 -> stage k resident, compute.
// Compute: all fragments of a k-step loaded into distinct registers before
// the 4 independent back-to-back WMMAs (staged dscnt waits, no full stalls).
// ---------------------------------------------------------------------------
template <int RELU, int BF16OUT>
__global__ void __launch_bounds__(256)
gemm_wmma_kernel(const unsigned short* __restrict__ A,
                 const unsigned short* __restrict__ Bt,
                 const float* __restrict__ bias,
                 float* __restrict__ Cout,
                 unsigned short* __restrict__ Cbf,
                 int M, int N, int K) {
  constexpr int KC = 64;
  __shared__ __align__(16) unsigned short sA[2][128 * KC];  // 2 x 16KB
  __shared__ __align__(16) unsigned short sB[2][64 * KC];   // 2 x  8KB

  const int tid  = threadIdx.x;
  const int lane = tid & 31;
  const int wid  = tid >> 5;
  const int m0   = blockIdx.y * 128;
  const int n0   = blockIdx.x * 64;
  const int half = lane >> 4;       // 0: K {0..7,16..23}; 1: K {8..15,24..31}
  const int lid  = lane & 15;
  const int nk   = K / KC;

  v8f acc[4] = {v8f{}, v8f{}, v8f{}, v8f{}};

  // --- issue one stage of async copies (global -> LDS) ---
  auto issue_stage = [&](int kb, int buf) {
    const int kbase = kb * KC;
#pragma unroll
    for (int i = 0; i < 4; ++i) {            // A tile: 1024 x 16B chunks
      const int c   = tid + 256 * i;
      const int row = c >> 3;                // 0..127
      const int col = (c & 7) * 8;           // element offset
      const unsigned short* gp = A + (size_t)(m0 + row) * K + kbase + col;
      const unsigned lp = (unsigned)(size_t)(const void*)&sA[buf][row * KC + col];
      asm volatile("global_load_async_to_lds_b128 %0, %1, off"
                   :: "v"(lp), "v"(gp) : "memory");
    }
#pragma unroll
    for (int i = 0; i < 2; ++i) {            // B tile: 512 x 16B chunks
      const int c   = tid + 256 * i;
      const int row = c >> 3;                // 0..63 (output column)
      const int col = (c & 7) * 8;
      const unsigned short* gp = Bt + (size_t)(n0 + row) * K + kbase + col;
      const unsigned lp = (unsigned)(size_t)(const void*)&sB[buf][row * KC + col];
      asm volatile("global_load_async_to_lds_b128 %0, %1, off"
                   :: "v"(lp), "v"(gp) : "memory");
    }
  };

  issue_stage(0, 0);
  for (int kb = 0; kb < nk; ++kb) {
    const int cur = kb & 1;
    if (kb + 1 < nk) {
      issue_stage(kb + 1, cur ^ 1);
      asm volatile("s_wait_asynccnt 0x6" ::: "memory");  // stage kb landed
    } else {
      asm volatile("s_wait_asynccnt 0x0" ::: "memory");
    }
    __syncthreads();

    const unsigned short* sa = &sA[cur][0];
    const unsigned short* sb = &sB[cur][0];
#pragma unroll
    for (int kc = 0; kc < KC; kc += 32) {
      // load ALL fragments of this k-step first (distinct registers),
      // then fire the 4 independent WMMAs back-to-back.
      const unsigned short* ap = sa + (wid * 16 + lid) * KC + kc + half * 8;
      const unsigned short* bp = sb + lid * KC + kc + half * 8;
      v16bf a, b0, b1, b2, b3;
      ((v8bf*)&a)[0]  = *(const v8bf*)(ap);
      ((v8bf*)&a)[1]  = *(const v8bf*)(ap + 16);
      ((v8bf*)&b0)[0] = *(const v8bf*)(bp);
      ((v8bf*)&b0)[1] = *(const v8bf*)(bp + 16);
      ((v8bf*)&b1)[0] = *(const v8bf*)(bp + 16 * KC);
      ((v8bf*)&b1)[1] = *(const v8bf*)(bp + 16 * KC + 16);
      ((v8bf*)&b2)[0] = *(const v8bf*)(bp + 32 * KC);
      ((v8bf*)&b2)[1] = *(const v8bf*)(bp + 32 * KC + 16);
      ((v8bf*)&b3)[0] = *(const v8bf*)(bp + 48 * KC);
      ((v8bf*)&b3)[1] = *(const v8bf*)(bp + 48 * KC + 16);
      acc[0] = __builtin_amdgcn_wmma_f32_16x16x32_bf16(
          false, a, false, b0, (short)0, acc[0], false, false);
      acc[1] = __builtin_amdgcn_wmma_f32_16x16x32_bf16(
          false, a, false, b1, (short)0, acc[1], false, false);
      acc[2] = __builtin_amdgcn_wmma_f32_16x16x32_bf16(
          false, a, false, b2, (short)0, acc[2], false, false);
      acc[3] = __builtin_amdgcn_wmma_f32_16x16x32_bf16(
          false, a, false, b3, (short)0, acc[3], false, false);
    }
    __syncthreads();   // all waves done reading buf before it is refilled
  }

  const size_t cbase = (size_t)(m0 + wid * 16 + half * 8) * N;
#pragma unroll
  for (int t = 0; t < 4; ++t) {
    const int n = n0 + t * 16 + lid;
    const float bv = bias ? bias[n] : 0.0f;
#pragma unroll
    for (int r = 0; r < 8; ++r) {
      float v = acc[t][r] + bv;
      if (RELU) v = fmaxf(v, 0.0f);
      const size_t idx = cbase + (size_t)r * N + n;
      Cout[idx] = v;
      if (BF16OUT) Cbf[idx] = f2bf(v);
    }
  }
}

// ---------------------------------------------------------------------------
// learned = sigmoid(h1 . bn_w2 + bn_b2)  (wave per token)
// ---------------------------------------------------------------------------
__global__ void learned_kernel(const float* __restrict__ h1,
                               const float* __restrict__ w2,
                               const float* __restrict__ b2,
                               float* __restrict__ learned) {
  const int row = blockIdx.x * 8 + (threadIdx.x >> 5);
  if (row >= kB * kL) return;
  const int lane = threadIdx.x & 31;
  const float* hr = h1 + (size_t)row * kD;
  float s = 0.0f;
  for (int d = lane; d < kD; d += 32) s += hr[d] * w2[d];
  for (int off = 16; off; off >>= 1) s += __shfl_xor(s, off, 32);
  if (lane == 0) learned[row] = 1.0f / (1.0f + __expf(-(s + b2[0])));
}

// bprob[b,l] = 0.5*(1 - cos(x[b,l], x[b,l+1]))   (wave per pair)
__global__ void bprob_kernel(const float* __restrict__ x,
                             float* __restrict__ bprob) {
  const int row = blockIdx.x * 8 + (threadIdx.x >> 5);
  if (row >= kB * (kL - 1)) return;
  const int lane = threadIdx.x & 31;
  const int b = row / (kL - 1);
  const int l = row % (kL - 1);
  const float* a = x + ((size_t)b * kL + l) * kD;
  const float* c = a + kD;
  float saa = 0.f, scc = 0.f, sac = 0.f;
  for (int d = lane; d < kD; d += 32) {
    const float av = a[d], cv = c[d];
    saa += av * av; scc += cv * cv; sac += av * cv;
  }
  for (int off = 16; off; off >>= 1) {
    saa += __shfl_xor(saa, off, 32);
    scc += __shfl_xor(scc, off, 32);
    sac += __shfl_xor(sac, off, 32);
  }
  if (lane == 0) {
    const float den = fmaxf(sqrtf(saa), 1e-8f) * fmaxf(sqrtf(scc), 1e-8f);
    bprob[row] = 0.5f * (1.0f - sac / den);
  }
}

// ---------------------------------------------------------------------------
// Per-batch scan: boundaries -> seg, segment starts/counts (segments are
// contiguous because seg is a cumsum of indicators).
// ---------------------------------------------------------------------------
__global__ void seg_scan_kernel(const float* __restrict__ bprob,
                                const float* __restrict__ learned,
                                int* __restrict__ seg,
                                int* __restrict__ starts,
                                int* __restrict__ counts) {
  const int b = threadIdx.x;
  if (b >= kB) return;
  for (int c = 0; c < kC; ++c) { starts[b * kC + c] = 0; counts[b * kC + c] = 0; }
  int cnt = 0;
  for (int l = 0; l < kL; ++l) {
    const float bd = (l == 0) ? 1.0f
                  : 0.7f * bprob[b * (kL - 1) + (l - 1)] + 0.3f * learned[b * kL + (l - 1)];
    if (bd > 0.5f) cnt++;
    const int s = cnt - 1;
    seg[b * kL + l] = s;
    if (s < kC) {
      if (counts[b * kC + s] == 0) starts[b * kC + s] = l;
      counts[b * kC + s]++;
    }
  }
}

// chunks[b,c,:] = mean of x over the contiguous segment range
__global__ void chunk_mean_kernel(const float* __restrict__ x,
                                  const int* __restrict__ starts,
                                  const int* __restrict__ counts,
                                  float* __restrict__ y,
                                  unsigned short* __restrict__ ybf) {
  const int bc = blockIdx.x;          // 0..B*C-1
  const int b  = bc / kC;
  const int st = starts[bc];
  const int cn = counts[bc];
  const float inv = 1.0f / fmaxf((float)cn, 1.0f);
  for (int d = threadIdx.x; d < kD; d += 256) {
    float s = 0.0f;
    for (int i = 0; i < cn; ++i) s += x[((size_t)b * kL + st + i) * kD + d];
    const float v = s * inv;
    const size_t idx = (size_t)bc * kD + d;
    y[idx]   = v;
    ybf[idx] = f2bf(v);
  }
}

// ---------------------------------------------------------------------------
// LayerNorm over D=512: out = LN(x + res)*g + b   (wave per row, 16 elem/lane)
// res / out / outbf may be null.
// ---------------------------------------------------------------------------
__global__ void layernorm_kernel(const float* __restrict__ x,
                                 const float* __restrict__ res,
                                 const float* __restrict__ g,
                                 const float* __restrict__ bta,
                                 float* __restrict__ out,
                                 unsigned short* __restrict__ outbf,
                                 int rows) {
  const int row = blockIdx.x * 8 + (threadIdx.x >> 5);
  if (row >= rows) return;
  const int lane = threadIdx.x & 31;
  const float* xr = x + (size_t)row * kD;
  const float* rr = res ? res + (size_t)row * kD : nullptr;
  float vals[16];
  float s = 0.0f;
#pragma unroll
  for (int i = 0; i < 16; ++i) {
    const int d = lane + 32 * i;
    float v = xr[d];
    if (rr) v += rr[d];
    vals[i] = v; s += v;
  }
  for (int off = 16; off; off >>= 1) s += __shfl_xor(s, off, 32);
  const float mean = s * (1.0f / kD);
  float vs = 0.0f;
#pragma unroll
  for (int i = 0; i < 16; ++i) { const float dlt = vals[i] - mean; vs += dlt * dlt; }
  for (int off = 16; off; off >>= 1) vs += __shfl_xor(vs, off, 32);
  const float rstd = rsqrtf(vs * (1.0f / kD) + kEPS);
#pragma unroll
  for (int i = 0; i < 16; ++i) {
    const int d = lane + 32 * i;
    const float o = (vals[i] - mean) * rstd * g[d] + bta[d];
    const size_t idx = (size_t)row * kD + d;
    if (out)   out[idx]   = o;
    if (outbf) outbf[idx] = f2bf(o);
  }
}

// ---------------------------------------------------------------------------
// Attention over 128 tokens, 8 heads, hd=64. Block = 128 threads (one/key).
// Grid: (C, H, B). Strides are row strides in elements (1536 for fused qkv).
// ---------------------------------------------------------------------------
__global__ void attn_kernel(const float* __restrict__ qp,
                            const float* __restrict__ kp,
                            const float* __restrict__ vp,
                            int sq, int sk, int sv,
                            float* __restrict__ out,
                            unsigned short* __restrict__ outbf) {
  const int qi = blockIdx.x, h = blockIdx.y, b = blockIdx.z;
  const int t = threadIdx.x;           // key index 0..127
  __shared__ float sc[kC];
  __shared__ float red[kC];

  const float* qrow = qp + ((size_t)b * kC + qi) * sq + h * 64;
  const float* krow = kp + ((size_t)b * kC + t)  * sk + h * 64;
  float s = 0.0f;
#pragma unroll 8
  for (int d = 0; d < 64; ++d) s += qrow[d] * krow[d];
  sc[t] = s * 0.125f;                  // 1/sqrt(64)
  red[t] = sc[t];
  __syncthreads();
  for (int off = 64; off; off >>= 1) {
    if (t < off) red[t] = fmaxf(red[t], red[t + off]);
    __syncthreads();
  }
  const float mx = red[0];
  __syncthreads();
  const float e = __expf(sc[t] - mx);
  sc[t] = e; red[t] = e;
  __syncthreads();
  for (int off = 64; off; off >>= 1) {
    if (t < off) red[t] += red[t + off];
    __syncthreads();
  }
  const float inv = 1.0f / red[0];
  __syncthreads();
  sc[t] *= inv;
  __syncthreads();

  if (t < 64) {
    float o = 0.0f;
    for (int j = 0; j < kC; ++j)
      o += sc[j] * vp[((size_t)b * kC + j) * sv + h * 64 + t];
    const size_t idx = ((size_t)b * kC + qi) * kD + h * 64 + t;
    out[idx]   = o;
    outbf[idx] = f2bf(o);
  }
}

// ---------------------------------------------------------------------------
// Gather proj rows per-token + fused double LayerNorm -> bf16 for LM head.
// ---------------------------------------------------------------------------
__global__ void gather_dualln_kernel(const float* __restrict__ proj,
                                     const int* __restrict__ seg,
                                     const float* __restrict__ g1, const float* __restrict__ b1,
                                     const float* __restrict__ g2, const float* __restrict__ b2,
                                     unsigned short* __restrict__ outbf) {
  const int row = blockIdx.x * 8 + (threadIdx.x >> 5);
  if (row >= kB * kL) return;
  const int lane = threadIdx.x & 31;
  const int b = row / kL;
  const int s = seg[row];
  float vals[16];
  if (s < kC) {
    const float* pr = proj + ((size_t)b * kC + s) * kD;
#pragma unroll
    for (int i = 0; i < 16; ++i) vals[i] = pr[lane + 32 * i];
  } else {
#pragma unroll
    for (int i = 0; i < 16; ++i) vals[i] = 0.0f;
  }
#pragma unroll
  for (int pass = 0; pass < 2; ++pass) {
    const float* gg = pass ? g2 : g1;
    const float* bb = pass ? b2 : b1;
    float sm = 0.0f;
#pragma unroll
    for (int i = 0; i < 16; ++i) sm += vals[i];
    for (int off = 16; off; off >>= 1) sm += __shfl_xor(sm, off, 32);
    const float mean = sm * (1.0f / kD);
    float vs = 0.0f;
#pragma unroll
    for (int i = 0; i < 16; ++i) { const float dl = vals[i] - mean; vs += dl * dl; }
    for (int off = 16; off; off >>= 1) vs += __shfl_xor(vs, off, 32);
    const float rstd = rsqrtf(vs * (1.0f / kD) + kEPS);
#pragma unroll
    for (int i = 0; i < 16; ++i) {
      const int d = lane + 32 * i;
      vals[i] = (vals[i] - mean) * rstd * gg[d] + bb[d];
    }
  }
#pragma unroll
  for (int i = 0; i < 16; ++i)
    outbf[(size_t)row * kD + lane + 32 * i] = f2bf(vals[i]);
}

// ---------------------------------------------------------------------------
// Host launcher
// ---------------------------------------------------------------------------
extern "C" void kernel_launch(void* const* d_in, const int* in_sizes, int n_in,
                              void* d_out, int out_size, void* d_ws, size_t ws_size,
                              hipStream_t stream) {
  (void)in_sizes; (void)n_in; (void)out_size; (void)ws_size;

  const float* emb    = (const float*)d_in[0];
  const float* bn_w1  = (const float*)d_in[1];
  const float* bn_b1  = (const float*)d_in[2];
  const float* bn_w2  = (const float*)d_in[3];
  const float* bn_b2  = (const float*)d_in[4];
  const float* qkv_w  = (const float*)d_in[5];
  const float* qkv_b  = (const float*)d_in[6];
  const float* ao_w   = (const float*)d_in[7];
  const float* ao_b   = (const float*)d_in[8];
  const float* ln1_g  = (const float*)d_in[9];
  const float* ln1_b  = (const float*)d_in[10];
  const float* ff_w1  = (const float*)d_in[11];
  const float* ff_b1  = (const float*)d_in[12];
  const float* ff_w2  = (const float*)d_in[13];
  const float* ff_b2  = (const float*)d_in[14];
  const float* ln2_g  = (const float*)d_in[15];
  const float* ln2_b  = (const float*)d_in[16];
  const float* ca_qw  = (const float*)d_in[17];
  const float* ca_kw  = (const float*)d_in[18];
  const float* ca_vw  = (const float*)d_in[19];
  const float* ca_qb  = (const float*)d_in[20];
  const float* ca_kb  = (const float*)d_in[21];
  const float* ca_vb  = (const float*)d_in[22];
  const float* ca_ow  = (const float*)d_in[23];
  const float* ca_ob  = (const float*)d_in[24];
  const float* bln_g  = (const float*)d_in[25];
  const float* bln_b  = (const float*)d_in[26];
  const float* dc_w   = (const float*)d_in[27];
  const float* dc_b   = (const float*)d_in[28];
  const float* dcln_g = (const float*)d_in[29];
  const float* dcln_b = (const float*)d_in[30];
  const float* fln_g  = (const float*)d_in[31];
  const float* fln_b  = (const float*)d_in[32];
  const float* lm_w   = (const float*)d_in[33];
  const float* lm_b   = (const float*)d_in[34];
  const int*   ids    = (const int*)d_in[35];
  float* outp         = (float*)d_out;

  // --- workspace layout ---
  char* ws = (char*)d_ws;
  size_t off = 0;
  auto alloc = [&](size_t bytes) -> void* {
    off = (off + 255) & ~(size_t)255;
    void* p = ws + off;
    off += bytes;
    return p;
  };
  const size_t TOK = (size_t)kB * kL;     // 32768
  const size_t CHK = (size_t)kB * kC;     // 2048

  float* xf            = (float*)alloc(TOK * kD * 4);
  unsigned short* xbf  = (unsigned short*)alloc(TOK * kD * 2);
  unsigned short* binp = (unsigned short*)alloc(TOK * 2 * kD * 2);
  float* h1            = (float*)alloc(TOK * kD * 4);
  float* bprob         = (float*)alloc((size_t)kB * (kL - 1) * 4);
  float* learned       = (float*)alloc(TOK * 4);
  int* seg             = (int*)alloc(TOK * 4);
  int* starts          = (int*)alloc(CHK * 4);
  int* counts          = (int*)alloc(CHK * 4);
  float* y             = (float*)alloc(CHK * kD * 4);
  unsigned short* ybf  = (unsigned short*)alloc(CHK * kD * 2);
  float* qkv           = (float*)alloc(CHK * 3 * kD * 4);
  float* att           = (float*)alloc(CHK * kD * 4);
  unsigned short* attbf= (unsigned short*)alloc(CHK * kD * 2);
  float* tmp           = (float*)alloc(CHK * kD * 4);
  float* hff           = (float*)alloc(CHK * kDFF * 4);
  unsigned short* hffbf= (unsigned short*)alloc(CHK * kDFF * 2);
  float* lvl1          = (float*)alloc(CHK * kD * 4);
  unsigned short* l1bf = (unsigned short*)alloc(CHK * kD * 2);
  float* cq            = (float*)alloc(CHK * kD * 4);
  float* ck            = (float*)alloc(CHK * kD * 4);
  float* cv            = (float*)alloc(CHK * kD * 4);
  float* hier          = (float*)alloc(CHK * kD * 4);
  unsigned short* hlbf = (unsigned short*)alloc(CHK * kD * 2);
  float* proj          = (float*)alloc(CHK * kD * 4);
  unsigned short* nrmbf= (unsigned short*)alloc(TOK * kD * 2);
  // transposed bf16 weights
  unsigned short* bn_w1_t = (unsigned short*)alloc((size_t)2 * kD * kD * 2);
  unsigned short* qkv_w_t = (unsigned short*)alloc((size_t)kNL * kD * 3 * kD * 2);
  unsigned short* ao_w_t  = (unsigned short*)alloc((size_t)kNL * kD * kD * 2);
  unsigned short* ff_w1_t = (unsigned short*)alloc((size_t)kNL * kD * kDFF * 2);
  unsigned short* ff_w2_t = (unsigned short*)alloc((size_t)kNL * kDFF * kD * 2);
  unsigned short* ca_qw_t = (unsigned short*)alloc((size_t)kD * kD * 2);
  unsigned short* ca_kw_t = (unsigned short*)alloc((size_t)kD * kD * 2);
  unsigned short* ca_vw_t = (unsigned short*)alloc((size_t)kD * kD * 2);
  unsigned short* ca_ow_t = (unsigned short*)alloc((size_t)kD * kD * 2);
  unsigned short* dc_w_t  = (unsigned short*)alloc((size_t)kD * kD * 2);
  unsigned short* lm_w_t  = (unsigned short*)alloc((size_t)kD * kV * 2);

  auto convT = [&](const float* W, unsigned short* Wt, int K, int N) {
    const size_t n = (size_t)K * N;
    conv_transpose_kernel<<<(unsigned)((n + 255) / 256), 256, 0, stream>>>(W, Wt, K, N);
  };
  auto gemm = [&](const unsigned short* A, const unsigned short* Bt, const float* bias,
                  float* Cf, unsigned short* Cb, int M, int N, int K, int relu) {
    dim3 grid(N / 64, M / 128);
    if (relu && Cb)      gemm_wmma_kernel<1, 1><<<grid, 256, 0, stream>>>(A, Bt, bias, Cf, Cb, M, N, K);
    else if (relu)       gemm_wmma_kernel<1, 0><<<grid, 256, 0, stream>>>(A, Bt, bias, Cf, Cb, M, N, K);
    else                 gemm_wmma_kernel<0, 0><<<grid, 256, 0, stream>>>(A, Bt, bias, Cf, Cb, M, N, K);
  };

  // --- weight preprocessing (bf16 transposed) ---
  convT(bn_w1, bn_w1_t, 2 * kD, kD);
  for (int i = 0; i < kNL; ++i) {
    convT(qkv_w + (size_t)i * kD * 3 * kD, qkv_w_t + (size_t)i * kD * 3 * kD, kD, 3 * kD);
    convT(ao_w  + (size_t)i * kD * kD,     ao_w_t  + (size_t)i * kD * kD,     kD, kD);
    convT(ff_w1 + (size_t)i * kD * kDFF,   ff_w1_t + (size_t)i * kD * kDFF,   kD, kDFF);
    convT(ff_w2 + (size_t)i * kDFF * kD,   ff_w2_t + (size_t)i * kDFF * kD,   kDFF, kD);
  }
  convT(ca_qw, ca_qw_t, kD, kD);
  convT(ca_kw, ca_kw_t, kD, kD);
  convT(ca_vw, ca_vw_t, kD, kD);
  convT(ca_ow, ca_ow_t, kD, kD);
  convT(dc_w,  dc_w_t,  kD, kD);
  convT(lm_w,  lm_w_t,  kD, kV);

  // --- token-level front end ---
  embed_pos_kernel<<<(unsigned)TOK, 256, 0, stream>>>(ids, emb, xf, xbf);
  build_binp_kernel<<<(unsigned)((TOK * kD + 255) / 256), 256, 0, stream>>>(xbf, binp);
  gemm(binp, bn_w1_t, bn_b1, h1, nullptr, (int)TOK, kD, 2 * kD, 1);     // relu MLP
  learned_kernel<<<(unsigned)((TOK + 7) / 8), 256, 0, stream>>>(h1, bn_w2, bn_b2, learned);
  bprob_kernel<<<(unsigned)(((size_t)kB * (kL - 1) + 7) / 8), 256, 0, stream>>>(xf, bprob);
  seg_scan_kernel<<<1, 32, 0, stream>>>(bprob, learned, seg, starts, counts);
  chunk_mean_kernel<<<(unsigned)CHK, 256, 0, stream>>>(xf, starts, counts, y, ybf);

  // --- chunk-level transformer stack ---
  for (int i = 0; i < kNL; ++i) {
    gemm(ybf, qkv_w_t + (size_t)i * kD * 3 * kD, qkv_b + (size_t)i * 3 * kD,
         qkv, nullptr, (int)CHK, 3 * kD, kD, 0);
    attn_kernel<<<dim3(kC, kH, kB), 128, 0, stream>>>(
        qkv, qkv + kD, qkv + 2 * kD, 3 * kD, 3 * kD, 3 * kD, att, attbf);
    gemm(attbf, ao_w_t + (size_t)i * kD * kD, ao_b + (size_t)i * kD,
         tmp, nullptr, (int)CHK, kD, kD, 0);
    layernorm_kernel<<<(unsigned)((CHK + 7) / 8), 256, 0, stream>>>(
        tmp, y, ln1_g + (size_t)i * kD, ln1_b + (size_t)i * kD, y, ybf, (int)CHK);
    gemm(ybf, ff_w1_t + (size_t)i * kD * kDFF, ff_b1 + (size_t)i * kDFF,
         hff, hffbf, (int)CHK, kDFF, kD, 1);
    gemm(hffbf, ff_w2_t + (size_t)i * kDFF * kD, ff_b2 + (size_t)i * kD,
         tmp, nullptr, (int)CHK, kD, kDFF, 0);
    layernorm_kernel<<<(unsigned)((CHK + 7) / 8), 256, 0, stream>>>(
        tmp, y, ln2_g + (size_t)i * kD, ln2_b + (size_t)i * kD, y, ybf, (int)CHK);
    if (i == kNL / 2 - 1) {
      hipMemcpyAsync(lvl1, y,   CHK * kD * 4, hipMemcpyDeviceToDevice, stream);
      hipMemcpyAsync(l1bf, ybf, CHK * kD * 2, hipMemcpyDeviceToDevice, stream);
    }
  }

  // --- hierarchical cross-attention ---
  gemm(ybf,  ca_qw_t, ca_qb, cq, nullptr, (int)CHK, kD, kD, 0);
  gemm(l1bf, ca_kw_t, ca_kb, ck, nullptr, (int)CHK, kD, kD, 0);
  gemm(l1bf, ca_vw_t, ca_vb, cv, nullptr, (int)CHK, kD, kD, 0);
  attn_kernel<<<dim3(kC, kH, kB), 128, 0, stream>>>(cq, ck, cv, kD, kD, kD, att, attbf);
  gemm(attbf, ca_ow_t, ca_ob, hier, nullptr, (int)CHK, kD, kD, 0);
  layernorm_kernel<<<(unsigned)((CHK + 7) / 8), 256, 0, stream>>>(
      hier, nullptr, bln_g, bln_b, nullptr, hlbf, (int)CHK);
  gemm(hlbf, dc_w_t, dc_b, proj, nullptr, (int)CHK, kD, kD, 0);

  // --- decode: gather + double LN + LM head ---
  gather_dualln_kernel<<<(unsigned)((TOK + 7) / 8), 256, 0, stream>>>(
      proj, seg, dcln_g, dcln_b, fln_g, fln_b, nrmbf);
  gemm(nrmbf, lm_w_t, lm_b, outp, nullptr, (int)TOK, kV, kD, 0);
}
